// SAGE_79877801771079
// MI455X (gfx1250) — matrix operand order repaired and checked by reading, not compiled, not run on hardware
//
#include <hip/hip_runtime.h>

// GraphSAGE-style: h = ReLU(x@W1^T + b); aggr = scatter-max over edges (+self);
// out = ReLU([aggr|x]@W2^T).  WMMA f16 for the GEMMs, u32 atomic-max scatter.

typedef __attribute__((ext_vector_type(16))) _Float16 v16h;
typedef __attribute__((ext_vector_type(8)))  float    v8f;

union FragU { v16h v; unsigned u[8]; };

// A fragment (16x32, f16) from LDS tile laid out [row][k], rowLen halves/row.
// ISA layout: lane m=lane&15, hs=lane>>4; VGPR v holds K pair
//   (v&3)*2 + hs*8 + (v>=4 ? 16 : 0)
__device__ __forceinline__ v16h ldsA(const _Float16* base, int rowLen, int lane, int kOff) {
  const unsigned* p = (const unsigned*)base;
  FragU t;
  const int m = lane & 15, hs = lane >> 4;
#pragma unroll
  for (int v = 0; v < 8; ++v) {
    int kb = (v & 3) * 2 + hs * 8 + ((v & 4) ? 16 : 0);
    t.u[v] = p[(m * rowLen + kOff + kb) >> 1];
  }
  return t.v;
}

// B fragment (32x16, f16) from LDS tile laid out [n][k] (i.e. W[c][k]).
// ISA layout: lane n=lane&15, hs=lane>>4; VGPR v holds K pair (v&3)*2 + hs*8 + (v>=4 ? 16 : 0).
__device__ __forceinline__ v16h ldsB(const _Float16* base, int rowLen, int lane, int kOff) {
  const unsigned* p = (const unsigned*)base;
  FragU t;
  const int nr = lane & 15, hs = lane >> 4;
#pragma unroll
  for (int v = 0; v < 8; ++v) {
    int kb = (v & 3) * 2 + hs * 8 + ((v & 4) ? 16 : 0);
    t.u[v] = p[(nr * rowLen + kOff + kb) >> 1];
  }
  return t.v;
}

// ---------------- Phase 1: h = ReLU(x @ lin_w^T + b); aggr = h (self-loop) ---
__global__ __launch_bounds__(128) void sage_gemm1(
    const float* __restrict__ x, const float* __restrict__ lin_w,
    const float* __restrict__ lin_b, float* __restrict__ h,
    float* __restrict__ aggr, int n) {
  __shared__ __align__(16) _Float16 Bs[128 * 128];      // W1 as f16, [c][k]
  __shared__ __align__(16) _Float16 Xs[4][16 * 128];    // per-wave 16-row x tile
  const int tid = threadIdx.x, lane = tid & 31, wv = tid >> 5;

  // fp32 -> fp16 weights into LDS (4096 float4 across 128 threads)
#pragma unroll
  for (int i = 0; i < 32; ++i) {
    int f = tid + 128 * i;
    float4 w4 = ((const float4*)lin_w)[f];
    int o = f * 4;
    Bs[o]   = (_Float16)w4.x; Bs[o+1] = (_Float16)w4.y;
    Bs[o+2] = (_Float16)w4.z; Bs[o+3] = (_Float16)w4.w;
  }
  const int r0 = blockIdx.x * 64 + wv * 16;
#pragma unroll
  for (int i = 0; i < 16; ++i) {
    int f = lane + 32 * i;          // 512 float4 per 16x128 tile
    int row = f >> 5, c4 = f & 31;
    float4 xv = make_float4(0.f, 0.f, 0.f, 0.f);
    if (r0 + row < n) xv = ((const float4*)(x + (size_t)(r0 + row) * 128))[c4];
    int o = row * 128 + c4 * 4;
    Xs[wv][o]   = (_Float16)xv.x; Xs[wv][o+1] = (_Float16)xv.y;
    Xs[wv][o+2] = (_Float16)xv.z; Xs[wv][o+3] = (_Float16)xv.w;
  }
  __syncthreads();

  v8f acc[8] = {};
#pragma unroll
  for (int kt = 0; kt < 4; ++kt) {
    v16h a = ldsA(Xs[wv], 128, lane, kt * 32);
#pragma unroll
    for (int nt = 0; nt < 8; ++nt) {
      v16h b = ldsB(Bs + nt * 16 * 128, 128, lane, kt * 32);
      acc[nt] = __builtin_amdgcn_wmma_f32_16x16x32_f16(
          false, a, false, b, (short)0, acc[nt], false, false);
    }
  }

  const int nlo = lane & 15, hs = lane >> 4;
#pragma unroll
  for (int nt = 0; nt < 8; ++nt) {
    int col = nt * 16 + nlo;
    float bias = lin_b[col];
#pragma unroll
    for (int r = 0; r < 8; ++r) {
      int row = r0 + r + 8 * hs;
      if (row < n) {
        float val = acc[nt][r] + bias;
        val = val > 0.f ? val : 0.f;
        h[(size_t)row * 128 + col]    = val;  // message source
        aggr[(size_t)row * 128 + col] = val;  // self-loop init for scatter-max
      }
    }
  }
}

// ---------------- Phase 2: scatter-max over edges (u32 max == f32 max, vals>=0)
__global__ __launch_bounds__(256) void sage_edges(
    const int* __restrict__ src, const int* __restrict__ dst,
    const float* __restrict__ h, unsigned* __restrict__ aggr, int ne) {
  const int lane = threadIdx.x & 31;
  const int wid  = (int)((blockIdx.x * blockDim.x + threadIdx.x) >> 5);
  const int nw   = (int)((gridDim.x * blockDim.x) >> 5);
  for (int e = wid; e < ne; e += nw) {
    int s = src[e], d = dst[e];
    if (e + nw < ne)  // gfx1250 global_prefetch_b8 for the next gather row
      __builtin_prefetch(h + (size_t)src[e + nw] * 128 + lane * 4, 0, 1);
    float4 m = ((const float4*)(h + (size_t)s * 128))[lane];
    unsigned* ap = aggr + (size_t)d * 128 + (size_t)lane * 4;
    atomicMax(ap + 0, __float_as_uint(m.x));
    atomicMax(ap + 1, __float_as_uint(m.y));
    atomicMax(ap + 2, __float_as_uint(m.z));
    atomicMax(ap + 3, __float_as_uint(m.w));
  }
}

// ---------------- Phase 3: out = ReLU([aggr | x] @ update_w^T) ----------------
__global__ __launch_bounds__(128) void sage_gemm2(
    const float* __restrict__ aggr, const float* __restrict__ x,
    const float* __restrict__ uw, float* __restrict__ out, int n) {
  __shared__ __align__(16) _Float16 Bs[128 * 128];
  __shared__ __align__(16) _Float16 Xs[4][16 * 128];
  const int tid = threadIdx.x, lane = tid & 31, wv = tid >> 5;
  const int r0 = blockIdx.x * 64 + wv * 16;

  v8f acc[8] = {};
#pragma unroll
  for (int half = 0; half < 2; ++half) {   // K=256 as two K=128 halves
    if (half) __syncthreads();             // protect LDS reuse
    const float* srcbuf = half ? x : aggr; // concat([aggr, x]) along K
#pragma unroll
    for (int i = 0; i < 32; ++i) {         // uw[c][half*128 + k] -> Bs[c][k]
      int f = tid + 128 * i;
      int c = f >> 5, c4 = f & 31;
      float4 w4 = ((const float4*)uw)[c * 64 + half * 32 + c4];
      int o = c * 128 + c4 * 4;
      Bs[o]   = (_Float16)w4.x; Bs[o+1] = (_Float16)w4.y;
      Bs[o+2] = (_Float16)w4.z; Bs[o+3] = (_Float16)w4.w;
    }
#pragma unroll
    for (int i = 0; i < 16; ++i) {
      int f = lane + 32 * i;
      int row = f >> 5, c4 = f & 31;
      float4 xv = make_float4(0.f, 0.f, 0.f, 0.f);
      if (r0 + row < n)
        xv = ((const float4*)(srcbuf + (size_t)(r0 + row) * 128))[c4];
      int o = row * 128 + c4 * 4;
      Xs[wv][o]   = (_Float16)xv.x; Xs[wv][o+1] = (_Float16)xv.y;
      Xs[wv][o+2] = (_Float16)xv.z; Xs[wv][o+3] = (_Float16)xv.w;
    }
    __syncthreads();
#pragma unroll
    for (int kt = 0; kt < 4; ++kt) {
      v16h a = ldsA(Xs[wv], 128, lane, kt * 32);
#pragma unroll
      for (int nt = 0; nt < 8; ++nt) {
        v16h b = ldsB(Bs + nt * 16 * 128, 128, lane, kt * 32);
        acc[nt] = __builtin_amdgcn_wmma_f32_16x16x32_f16(
            false, a, false, b, (short)0, acc[nt], false, false);
      }
    }
  }

  const int nlo = lane & 15, hs = lane >> 4;
#pragma unroll
  for (int nt = 0; nt < 8; ++nt) {
    int col = nt * 16 + nlo;
#pragma unroll
    for (int r = 0; r < 8; ++r) {
      int row = r0 + r + 8 * hs;
      if (row < n) {
        float val = acc[nt][r];
        out[(size_t)row * 128 + col] = val > 0.f ? val : 0.f;
      }
    }
  }
}

extern "C" void kernel_launch(void* const* d_in, const int* in_sizes, int n_in,
                              void* d_out, int out_size, void* d_ws, size_t ws_size,
                              hipStream_t stream) {
  (void)n_in; (void)out_size; (void)ws_size;
  const float* x        = (const float*)d_in[0];
  const int*   ei       = (const int*)d_in[1];   // [2, E]: src then dst (harness: int)
  const float* lin_w    = (const float*)d_in[2];
  const float* lin_b    = (const float*)d_in[3];
  const float* update_w = (const float*)d_in[4];
  float* out = (float*)d_out;

  const int n  = in_sizes[0] / 128;  // 50000
  const int ne = in_sizes[1] / 2;    // 600000

  float* h    = (float*)d_ws;                 // [n,128] messages
  float* aggr = h + (size_t)n * 128;          // [n,128] scatter-max result

  const int gblocks = (n + 63) / 64;          // 64 rows per workgroup (4 waves x 16)
  sage_gemm1<<<gblocks, 128, 0, stream>>>(x, lin_w, lin_b, h, aggr, n);
  sage_edges<<<1024, 256, 0, stream>>>(ei, ei + ne, h, (unsigned*)aggr, ne);
  sage_gemm2<<<gblocks, 128, 0, stream>>>(aggr, x, update_w, out, n);
}
